// GNNLayer_72430328480187
// MI455X (gfx1250) — compile-verified, matrix operand order checked by reading, not run on hardware
//
#include <hip/hip_runtime.h>

// ---------------------------------------------------------------------------
// GNN layer on MI455X (gfx1250): all GEMMs via v_wmma_f32_16x16x32_bf16.
//   edge MLP:   m = relu([e|hs|hr] @ W1 + b1) @ W2 + b2          (K=192,128)
//   agg:        atomic f32 scatter-add of m into receivers
//   edge upd:   e_new = relu([e|m] @ We + be)                    (K=128)
//   node upd:   h_new = relu([h|agg] @ Wn + bn)                  (K=128)
// ---------------------------------------------------------------------------

typedef __attribute__((ext_vector_type(16))) __bf16 v16bf;
typedef __attribute__((ext_vector_type(8)))  __bf16 v8bf;
typedef __attribute__((ext_vector_type(8)))  float  v8f;

#define D_NODE 64
#define D_HID  128
#define NW     8     // waves per block in the edge kernel

// ---- WMMA wrapper -----------------------------------------------------------
__device__ __forceinline__ v8f wmma_bf16(v16bf a, v16bf b, v8f c) {
  return __builtin_amdgcn_wmma_f32_16x16x32_bf16(
      /*neg_a=*/false, a, /*neg_b=*/false, b,
      /*c_mod=*/(short)0, c, /*reuse_a=*/false, /*reuse_b=*/false);
}

// ---- A-fragment builders ----------------------------------------------------
// 16-bit A-matrix 16x32 layout (ISA 7.12.2): lane(hi=l>>4, m=l&15) holds row m,
// elements 0..7 -> K = hi*8+0..7 ; elements 8..15 -> K = 16+hi*8+0..7.

__device__ __forceinline__ void load8_g(const float* __restrict__ src, v16bf& a, int at) {
  float4 x = *(const float4*)(src);
  float4 y = *(const float4*)(src + 4);
  a[at + 0] = (__bf16)x.x; a[at + 1] = (__bf16)x.y;
  a[at + 2] = (__bf16)x.z; a[at + 3] = (__bf16)x.w;
  a[at + 4] = (__bf16)y.x; a[at + 5] = (__bf16)y.y;
  a[at + 6] = (__bf16)y.z; a[at + 7] = (__bf16)y.w;
}

// One 16x32 K-chunk of an A matrix from a row-major f32 row (cb = chunk base col)
__device__ __forceinline__ v16bf make_a_frag(const float* __restrict__ row, int cb, int hi) {
  v16bf a;
  load8_g(row + cb + hi * 8, a, 0);
  load8_g(row + cb + 16 + hi * 8, a, 8);
  return a;
}

// Same, sourcing a bf16 row in LDS (8 contiguous bf16 = one b128 load)
__device__ __forceinline__ void load8_l(const __bf16* p, v16bf& a, int at) {
  v8bf v = *(const v8bf*)p;
#pragma unroll
  for (int j = 0; j < 8; ++j) a[at + j] = v[j];
}
__device__ __forceinline__ v16bf make_a_frag_lds(const __bf16* row, int cb, int hi) {
  v16bf a;
  load8_l(row + cb + hi * 8, a, 0);
  load8_l(row + cb + 16 + hi * 8, a, 8);
  return a;
}

// ---- B-fragments ------------------------------------------------------------
// Weights stored in LDS pre-swizzled per 16x16x32 fragment: fragment f occupies
// 512 bf16; lane l holds its 16 elements contiguously at f*512 + l*16.
__device__ __forceinline__ v16bf bfrag(const __bf16* base, int frag, int lane) {
  const __bf16* p = base + frag * 512 + lane * 16;
  v16bf b;
  load8_l(p, b, 0);
  load8_l(p + 8, b, 8);
  return b;
}

// Pre-swizzle a KxN f32 weight matrix into B-fragment order (bf16) in LDS.
// B 32x16 chunk layout mirrors the A layout with lane -> column:
//   lane(hi, n): element i -> K = kc*32 + (i<8 ? hi*8+i : 16+hi*8+(i-8)), col = nt*16+n
__device__ __forceinline__ void fill_wfrags(__bf16* dst, const float* __restrict__ W,
                                            int K, int N) {
  const int nNT = N >> 4;
  const int total = K * N;
  for (int t = threadIdx.x; t < total; t += blockDim.x) {
    int frag = t >> 9;
    int s    = t & 511;
    int lane = s >> 4;
    int i    = s & 15;
    int kc   = frag / nNT;
    int nt   = frag - kc * nNT;
    int hi   = lane >> 4;
    int n    = lane & 15;
    int k    = kc * 32 + ((i < 8) ? (hi * 8 + i) : (16 + hi * 8 + (i - 8)));
    int col  = nt * 16 + n;
    dst[t] = (__bf16)W[k * N + col];   // dst index == frag*512 + lane*16 + i == t
  }
}

// ---------------------------------------------------------------------------
// Kernel 0: zero the aggregation workspace
// ---------------------------------------------------------------------------
__global__ void gnn_zero_kernel(float* __restrict__ p, size_t n) {
  size_t i = (size_t)blockIdx.x * blockDim.x + threadIdx.x;
  size_t stride = (size_t)gridDim.x * blockDim.x;
  for (; i < n; i += stride) p[i] = 0.0f;
}

// ---------------------------------------------------------------------------
// Kernel 1: edge pipeline. One wave <-> one 16-edge tile per iteration.
// ---------------------------------------------------------------------------
__global__ __launch_bounds__(32 * NW) void gnn_edge_kernel(
    const float* __restrict__ h, const float* __restrict__ e,
    const int* __restrict__ senders, const int* __restrict__ receivers,
    const float* __restrict__ W1, const float* __restrict__ b1,
    const float* __restrict__ W2, const float* __restrict__ b2,
    const float* __restrict__ We, const float* __restrict__ be,
    float* __restrict__ e_new, float* __restrict__ agg, int n_edges) {
  // W1: 192x128 -> 6 K-chunks x 8 N-tiles = 48 frags; W2/We: 128x64 -> 16 frags
  __shared__ __attribute__((aligned(32))) __bf16 sW1[48 * 512];
  __shared__ __attribute__((aligned(32))) __bf16 sW2[16 * 512];
  __shared__ __attribute__((aligned(32))) __bf16 sWe[16 * 512];
  __shared__ float sb1[D_HID];
  __shared__ float sb2[D_NODE];
  __shared__ float sbe[D_NODE];
  // Per-wave staging (row pitches padded: 136 & 72 elems -> conflict-free b128s)
  __shared__ __attribute__((aligned(32))) __bf16 sH[NW][16][136];
  __shared__ __attribute__((aligned(32))) __bf16 sM[NW][16][72];

  fill_wfrags(sW1, W1, 192, 128);
  fill_wfrags(sW2, W2, 128, 64);
  fill_wfrags(sWe, We, 128, 64);
  for (int i = threadIdx.x; i < D_HID; i += blockDim.x) sb1[i] = b1[i];
  for (int i = threadIdx.x; i < D_NODE; i += blockDim.x) { sb2[i] = b2[i]; sbe[i] = be[i]; }
  __syncthreads();

  const int lane = threadIdx.x & 31;
  const int wid  = threadIdx.x >> 5;
  const int hi   = lane >> 4;   // half-wave
  const int mr   = lane & 15;   // A-row / BC-column within tile
  const int n_tiles = n_edges >> 4;   // N_EDGES is a multiple of 16

  for (int tile = blockIdx.x * NW + wid; tile < n_tiles; tile += gridDim.x * NW) {
    const int base = tile << 4;
    const int edge = base + mr;
    const int sidx = senders[edge];
    const int ridx = receivers[edge];
    const float* erow = e + (size_t)edge * D_NODE;
    const float* srow = h + (size_t)sidx * D_NODE;
    const float* rrow = h + (size_t)ridx * D_NODE;

    // A = [e | hs | hr] : 6 K-chunks of 32 (each chunk inside one segment)
    v16bf a0 = make_a_frag(erow, 0, hi);
    v16bf a1 = make_a_frag(erow, 32, hi);
    v16bf a2 = make_a_frag(srow, 0, hi);
    v16bf a3 = make_a_frag(srow, 32, hi);
    v16bf a4 = make_a_frag(rrow, 0, hi);
    v16bf a5 = make_a_frag(rrow, 32, hi);

    // ---- H = relu(A @ W1 + b1) : 16x128 ----
#pragma unroll
    for (int nt = 0; nt < 8; ++nt) {
      const float bv = sb1[nt * 16 + mr];
      v8f c = {bv, bv, bv, bv, bv, bv, bv, bv};
      c = wmma_bf16(a0, bfrag(sW1, 0 * 8 + nt, lane), c);
      c = wmma_bf16(a1, bfrag(sW1, 1 * 8 + nt, lane), c);
      c = wmma_bf16(a2, bfrag(sW1, 2 * 8 + nt, lane), c);
      c = wmma_bf16(a3, bfrag(sW1, 3 * 8 + nt, lane), c);
      c = wmma_bf16(a4, bfrag(sW1, 4 * 8 + nt, lane), c);
      c = wmma_bf16(a5, bfrag(sW1, 5 * 8 + nt, lane), c);
      // C-layout lane(hi,n) row r2 -> M = hi*8+r2 ; relu + bf16 -> LDS
#pragma unroll
      for (int r2 = 0; r2 < 8; ++r2) {
        float v = c[r2];
        v = v > 0.0f ? v : 0.0f;
        sH[wid][hi * 8 + r2][nt * 16 + mr] = (__bf16)v;
      }
    }

    // ---- M = H @ W2 + b2 : 16x64 (no relu) ----
    v16bf ha[4];
#pragma unroll
    for (int kc = 0; kc < 4; ++kc)
      ha[kc] = make_a_frag_lds(&sH[wid][mr][0], kc * 32, hi);

    int rcv[8];
#pragma unroll
    for (int r2 = 0; r2 < 8; ++r2) rcv[r2] = receivers[base + hi * 8 + r2];

#pragma unroll
    for (int nt = 0; nt < 4; ++nt) {
      const float bv = sb2[nt * 16 + mr];
      v8f c = {bv, bv, bv, bv, bv, bv, bv, bv};
#pragma unroll
      for (int kc = 0; kc < 4; ++kc)
        c = wmma_bf16(ha[kc], bfrag(sW2, kc * 4 + nt, lane), c);
#pragma unroll
      for (int r2 = 0; r2 < 8; ++r2) {
        const float mv = c[r2];
        sM[wid][hi * 8 + r2][nt * 16 + mr] = (__bf16)mv;  // for edge update
        // segment_sum: f32 atomic scatter-add into receiver rows
        atomicAdd(&agg[(size_t)rcv[r2] * D_NODE + nt * 16 + mr], mv);
      }
    }

    // ---- e_new = relu([e | M] @ We + be) : 16x64 ----
    v16bf m0 = make_a_frag_lds(&sM[wid][mr][0], 0, hi);
    v16bf m1 = make_a_frag_lds(&sM[wid][mr][0], 32, hi);
#pragma unroll
    for (int nt = 0; nt < 4; ++nt) {
      const float bv = sbe[nt * 16 + mr];
      v8f c = {bv, bv, bv, bv, bv, bv, bv, bv};
      c = wmma_bf16(a0, bfrag(sWe, 0 * 4 + nt, lane), c);
      c = wmma_bf16(a1, bfrag(sWe, 1 * 4 + nt, lane), c);
      c = wmma_bf16(m0, bfrag(sWe, 2 * 4 + nt, lane), c);
      c = wmma_bf16(m1, bfrag(sWe, 3 * 4 + nt, lane), c);
#pragma unroll
      for (int r2 = 0; r2 < 8; ++r2) {
        float v = c[r2];
        v = v > 0.0f ? v : 0.0f;
        e_new[(size_t)(base + hi * 8 + r2) * D_NODE + nt * 16 + mr] = v;
      }
    }
  }
}

// ---------------------------------------------------------------------------
// Kernel 2: node update h_new = relu([h | agg] @ Wn + bn)
// ---------------------------------------------------------------------------
__global__ __launch_bounds__(32 * NW) void gnn_node_kernel(
    const float* __restrict__ h, const float* __restrict__ agg,
    const float* __restrict__ Wn, const float* __restrict__ bn,
    float* __restrict__ h_new, int n_nodes) {
  __shared__ __attribute__((aligned(32))) __bf16 sWn[16 * 512];
  __shared__ float sbn[D_NODE];

  fill_wfrags(sWn, Wn, 128, 64);
  for (int i = threadIdx.x; i < D_NODE; i += blockDim.x) sbn[i] = bn[i];
  __syncthreads();

  const int lane = threadIdx.x & 31;
  const int wid  = threadIdx.x >> 5;
  const int hi   = lane >> 4;
  const int mr   = lane & 15;
  const int n_tiles = n_nodes >> 4;   // N_NODES multiple of 16

  for (int tile = blockIdx.x * NW + wid; tile < n_tiles; tile += gridDim.x * NW) {
    const int base = tile << 4;
    const int node = base + mr;
    const float* hrow = h + (size_t)node * D_NODE;
    const float* arow = agg + (size_t)node * D_NODE;

    v16bf a0 = make_a_frag(hrow, 0, hi);
    v16bf a1 = make_a_frag(hrow, 32, hi);
    v16bf a2 = make_a_frag(arow, 0, hi);
    v16bf a3 = make_a_frag(arow, 32, hi);

#pragma unroll
    for (int nt = 0; nt < 4; ++nt) {
      const float bv = sbn[nt * 16 + mr];
      v8f c = {bv, bv, bv, bv, bv, bv, bv, bv};
      c = wmma_bf16(a0, bfrag(sWn, 0 * 4 + nt, lane), c);
      c = wmma_bf16(a1, bfrag(sWn, 1 * 4 + nt, lane), c);
      c = wmma_bf16(a2, bfrag(sWn, 2 * 4 + nt, lane), c);
      c = wmma_bf16(a3, bfrag(sWn, 3 * 4 + nt, lane), c);
#pragma unroll
      for (int r2 = 0; r2 < 8; ++r2) {
        float v = c[r2];
        v = v > 0.0f ? v : 0.0f;
        h_new[(size_t)(base + hi * 8 + r2) * D_NODE + nt * 16 + mr] = v;
      }
    }
  }
}

// ---------------------------------------------------------------------------
extern "C" void kernel_launch(void* const* d_in, const int* in_sizes, int n_in,
                              void* d_out, int out_size, void* d_ws, size_t ws_size,
                              hipStream_t stream) {
  const float* h        = (const float*)d_in[0];
  const float* e        = (const float*)d_in[1];
  const int*   senders  = (const int*)d_in[2];
  const int*   receivers= (const int*)d_in[3];
  const float* W1       = (const float*)d_in[4];
  const float* b1       = (const float*)d_in[5];
  const float* W2       = (const float*)d_in[6];
  const float* b2       = (const float*)d_in[7];
  const float* Wn       = (const float*)d_in[8];
  const float* bn       = (const float*)d_in[9];
  const float* We       = (const float*)d_in[10];
  const float* be       = (const float*)d_in[11];

  const int n_edges = in_sizes[2];
  const int n_nodes = in_sizes[0] / D_NODE;

  float* h_new = (float*)d_out;                          // [n_nodes, 64]
  float* e_new = (float*)d_out + (size_t)n_nodes * D_NODE; // [n_edges, 64]
  float* agg   = (float*)d_ws;                           // [n_nodes, 64] f32

  const size_t n_agg = (size_t)n_nodes * D_NODE;
  gnn_zero_kernel<<<1024, 256, 0, stream>>>(agg, n_agg);

  const int edge_tiles = n_edges >> 4;
  int egrid = (edge_tiles + NW - 1) / NW;
  if (egrid > 1536) egrid = 1536;
  gnn_edge_kernel<<<egrid, 32 * NW, 0, stream>>>(
      h, e, senders, receivers, W1, b1, W2, b2, We, be, e_new, agg, n_edges);

  const int node_tiles = n_nodes >> 4;
  int ngrid = (node_tiles + NW - 1) / NW;
  if (ngrid > 1536) ngrid = 1536;
  gnn_node_kernel<<<ngrid, 32 * NW, 0, stream>>>(h, agg, Wn, bn, h_new, n_nodes);
}
